// CRLI_33036888441400
// MI455X (gfx1250) — compile-verified, matrix operand order checked by reading, not run on hardware
//
#include <hip/hip_runtime.h>
#include <hip/hip_bf16.h>

typedef unsigned short u16;
typedef __attribute__((ext_vector_type(16))) __bf16 v16bf;
typedef __attribute__((ext_vector_type(8)))  float  v8f;
typedef __attribute__((ext_vector_type(8)))  unsigned short v8us;

#define B_  512
#define S_  64
#define D_  8
#define H_  512
#define G4  2048   /* 4*H */

static __device__ __forceinline__ float sigmoidf_(float x) {
  return 1.0f / (1.0f + __expf(-x));
}
static __device__ __forceinline__ u16 f2bf(float x) {
  unsigned int u = __float_as_uint(x);
  unsigned int r = (u + 0x7FFFu + ((u >> 16) & 1u)) >> 16;
  return (u16)r;
}

union FragCvt { v8us u[2]; v16bf v; };

// A-frag (16x32 bf16): lane L<16 holds row M=L, K chunks {0..7} and {16..23};
// lane L>=16 holds row M=L-16, K chunks {8..15} and {24..31}.
// B-frag mirrors this with per-lane *columns* of B == rows of W (B = W^T).
static __device__ __forceinline__ v16bf load_row_frag(const u16* rowp, int kbase, int lane) {
  int sel = (lane >> 4) << 3;
  FragCvt f;
  f.u[0] = *(const v8us*)(rowp + kbase + sel);
  f.u[1] = *(const v8us*)(rowp + kbase + 16 + sel);
  return f.v;
}

static __device__ __forceinline__ v8f wmma_bf(v16bf a, v16bf b, v8f c) {
  return __builtin_amdgcn_wmma_f32_16x16x32_bf16(false, a, false, b, (short)0, c, false, false);
}

// ---------------- prep kernels ----------------

// Wcat1[r, 0:512] = Wih1[r,:], Wcat1[r, 512:1024] = Whh1[r,:]  (bf16)
__global__ void k_cat1(u16* dst, const float* Wih, const float* Whh) {
  int i = blockIdx.x * blockDim.x + threadIdx.x;
  if (i >= G4 * 1024) return;
  int r = i >> 10, k = i & 1023;
  float v = (k < H_) ? Wih[r * H_ + k] : Whh[r * H_ + (k - H_)];
  dst[i] = f2bf(v);
}

__global__ void k_conv_bf(u16* dst, const float* src, int n) {
  int i = blockIdx.x * blockDim.x + threadIdx.x;
  if (i < n) dst[i] = f2bf(src[i]);
}

// Wih0 [2048,8] -> padded [2048,32] bf16
__global__ void k_pad0(u16* dst, const float* Wih0) {
  int i = blockIdx.x * blockDim.x + threadIdx.x;
  if (i >= G4 * 32) return;
  int r = i >> 5, k = i & 31;
  dst[i] = (k < D_) ? f2bf(Wih0[r * D_ + k]) : (u16)0;
}

__global__ void k_addbias(float* dst, const float* a, const float* b, int n) {
  int i = blockIdx.x * blockDim.x + threadIdx.x;
  if (i < n) dst[i] = a[i] + b[i];
}

__global__ void k_zero16(u16* dst, int n) {
  int i = blockIdx.x * blockDim.x + threadIdx.x;
  if (i < n) dst[i] = 0;
}

// warmup: lstm_step(layer1, zeros, zeros, zeros) -> g = bih+bhh (already fused)
__global__ void k_warm(const float* b1f, const float* b1b, float* warmH, float* warmC) {
  int i = blockIdx.x * blockDim.x + threadIdx.x;
  if (i >= 2 * H_) return;
  int dir = i >> 9, n = i & (H_ - 1);
  const float* b1 = dir ? b1b : b1f;
  float gi = b1[n], gg = b1[2 * H_ + n], go = b1[3 * H_ + n];
  float c = sigmoidf_(gi) * tanhf(gg);          // f-term is *0
  float h = sigmoidf_(go) * tanhf(c);
  warmH[i] = h; warmC[i] = c;
}

__global__ void k_init(const float* warmH, const float* warmC,
                       u16* X1f, u16* X1b, float* C0f, float* C0b,
                       float* C1f, float* C1b, float* Hf, float* Hb) {
  int i = blockIdx.x * blockDim.x + threadIdx.x;
  if (i >= 2 * B_ * H_) return;
  int dir = i >> 18, r = i & ((1 << 18) - 1);
  int b = r >> 9, n = r & (H_ - 1);
  u16* X1 = dir ? X1b : X1f;
  float wh = warmH[dir * H_ + n], wc = warmC[dir * H_ + n];
  X1[b * 1024 + n] = 0;                 // h0 = 0
  X1[b * 1024 + H_ + n] = f2bf(wh);     // h1 = warmup
  (dir ? C0b : C0f)[b * H_ + n] = 0.0f;
  (dir ? C1b : C1f)[b * H_ + n] = wc;
  (dir ? Hb : Hf)[b * H_ + n] = wh;     // fp32 copy of top hidden (for ximp)
}

// ---------------- per-step kernels ----------------

// ximp = gen_out(h_top); in = (1-m)*ximp + m*x; write bf16 into inpad[:, 0:8]
__global__ void k_ximp(const float* Hf, const float* Hb, const float* Wgo, const float* bgo,
                       const float* values, const int* masks,
                       u16* inF, u16* inB, int t) {
  int i = blockIdx.x * blockDim.x + threadIdx.x;
  if (i >= 2 * B_ * D_) return;
  int dir = i >> 12, r = i & 4095;
  int b = r >> 3, d = r & 7;
  const float* Ht = dir ? Hb : Hf;
  float acc = bgo[d];
  const float* hr = Ht + b * H_;
  const float* wr = Wgo + d * H_;
  for (int k = 0; k < H_; ++k) acc += hr[k] * wr[k];
  int te = dir ? (S_ - 1 - t) : t;
  int idx = (b * S_ + te) * D_ + d;
  float m = (float)masks[idx];
  float in = (1.0f - m) * acc + m * values[idx];
  (dir ? inB : inF)[b * 32 + d] = f2bf(in);
}

// layer0 gates = inpad[B,32] @ Wih0p^T + h0[B,512] @ Whh0^T   (no bias)
__global__ void __launch_bounds__(256)
k_gemm_l0(const u16* inF, const u16* inB, const u16* W0F, const u16* W0B,
          const u16* X1f, const u16* X1b, const u16* WhF, const u16* WhB,
          float* Gf, float* Gb) {
  int dir = blockIdx.z;
  const u16* IN = dir ? inB : inF;   // [512,32]
  const u16* W0 = dir ? W0B : W0F;   // [2048,32]
  const u16* X  = dir ? X1b : X1f;   // [512,1024] (cols 0..511 = h0)
  const u16* WH = dir ? WhB : WhF;   // [2048,512]
  float* G = dir ? Gb : Gf;          // [512,2048]
  int lane = threadIdx.x & 31, wave = threadIdx.x >> 5;
  int m0 = (blockIdx.x * 2 + (wave >> 2)) * 16;
  int n0 = (blockIdx.y * 4 + (wave & 3)) * 16;
  int ar = m0 + (lane & 15), br = n0 + (lane & 15);
  v8f acc = {};
  acc = wmma_bf(load_row_frag(IN + ar * 32, 0, lane),
                load_row_frag(W0 + br * 32, 0, lane), acc);
  const u16* arow = X + ar * 1024;
  const u16* brow = WH + br * H_;
  #pragma unroll
  for (int k0 = 0; k0 < H_; k0 += 32)
    acc = wmma_bf(load_row_frag(arow, k0, lane), load_row_frag(brow, k0, lane), acc);
  int mr = m0 + ((lane >> 4) << 3);
  int nc = n0 + (lane & 15);
  #pragma unroll
  for (int j = 0; j < 8; ++j) G[(mr + j) * G4 + nc] = acc[j];
}

// layer1 gates = [h0 | h1][B,1024] @ Wcat1^T   (no bias)
__global__ void __launch_bounds__(256)
k_gemm_l1(const u16* X1f, const u16* X1b, const u16* WcF, const u16* WcB,
          float* Gf, float* Gb) {
  int dir = blockIdx.z;
  const u16* X = dir ? X1b : X1f;    // [512,1024]
  const u16* W = dir ? WcB : WcF;    // [2048,1024]
  float* G = dir ? Gb : Gf;
  int lane = threadIdx.x & 31, wave = threadIdx.x >> 5;
  int m0 = (blockIdx.x * 2 + (wave >> 2)) * 16;
  int n0 = (blockIdx.y * 4 + (wave & 3)) * 16;
  const u16* arow = X + (m0 + (lane & 15)) * 1024;
  const u16* brow = W + (n0 + (lane & 15)) * 1024;
  v8f acc = {};
  #pragma unroll 8
  for (int k0 = 0; k0 < 1024; k0 += 32)
    acc = wmma_bf(load_row_frag(arow, k0, lane), load_row_frag(brow, k0, lane), acc);
  int mr = m0 + ((lane >> 4) << 3);
  int nc = n0 + (lane & 15);
  #pragma unroll
  for (int j = 0; j < 8; ++j) G[(mr + j) * G4 + nc] = acc[j];
}

// pointwise LSTM update; writes new h (bf16) into X1[:, col_off:col_off+512]
__global__ void k_point(const float* Gf, const float* Gb, const float* bf, const float* bb,
                        float* Cf, float* Cb, u16* X1f, u16* X1b,
                        float* Hf, float* Hb, int col_off, int write_h32) {
  int i = blockIdx.x * blockDim.x + threadIdx.x;
  if (i >= 2 * B_ * H_) return;
  int dir = i >> 18, r = i & ((1 << 18) - 1);
  int b = r >> 9, n = r & (H_ - 1);
  const float* G = dir ? Gb : Gf;
  const float* bias = dir ? bb : bf;
  float* C = dir ? Cb : Cf;
  u16* X1 = dir ? X1b : X1f;
  float gi = G[b * G4 + n]            + bias[n];
  float gf = G[b * G4 + H_ + n]       + bias[H_ + n];
  float gg = G[b * G4 + 2 * H_ + n]   + bias[2 * H_ + n];
  float go = G[b * G4 + 3 * H_ + n]   + bias[3 * H_ + n];
  float c = sigmoidf_(gf) * C[b * H_ + n] + sigmoidf_(gi) * tanhf(gg);
  float h = sigmoidf_(go) * tanhf(c);
  C[b * H_ + n] = c;
  X1[b * 1024 + col_off + n] = f2bf(h);
  if (write_h32) (dir ? Hb : Hf)[b * H_ + n] = h;
}

// ---------------- post kernels ----------------

// h = (hf + flip(hb, hidden))/2 ; write impute; imputed for disc; bf16 h for fc
__global__ void k_hmean(const float* Hf, const float* Hb, const float* values, const int* masks,
                        float* out_impute, float* imputed, u16* hbf) {
  int i = blockIdx.x * blockDim.x + threadIdx.x;
  if (i >= B_ * H_) return;
  int b = i >> 9, n = i & (H_ - 1);
  float h = 0.5f * (Hf[b * H_ + n] + Hb[b * H_ + (H_ - 1 - n)]);
  out_impute[i] = h;
  float m = (float)masks[i];
  imputed[i] = h * (1.0f - m) + values[i] * m;
  hbf[i] = f2bf(h);
}

// latent = h @ fcW^T + fcb  (M=512,N=512,K=512)
__global__ void __launch_bounds__(256)
k_gemm_fc(const u16* hbf, const u16* Wfc, const float* fcb, float* lat) {
  int lane = threadIdx.x & 31, wave = threadIdx.x >> 5;
  int m0 = (blockIdx.x * 2 + (wave >> 2)) * 16;
  int n0 = (blockIdx.y * 4 + (wave & 3)) * 16;
  const u16* arow = hbf + (m0 + (lane & 15)) * H_;
  const u16* brow = Wfc + (n0 + (lane & 15)) * H_;
  v8f acc = {};
  #pragma unroll
  for (int k0 = 0; k0 < H_; k0 += 32)
    acc = wmma_bf(load_row_frag(arow, k0, lane), load_row_frag(brow, k0, lane), acc);
  int mr = m0 + ((lane >> 4) << 3);
  int nc = n0 + (lane & 15);
  #pragma unroll
  for (int j = 0; j < 8; ++j) lat[(mr + j) * H_ + nc] = acc[j] + fcb[nc];
}

// discriminator: zero-state LSTM cells -> Whh terms vanish
static __device__ void disc_cell(const float* Wih, const float* bih, const float* bhh,
                                 const float* x, int in_dim, int hid, float* out) {
  for (int n = 0; n < hid; ++n) {
    float gi = bih[n] + bhh[n];
    float gg = bih[2 * hid + n] + bhh[2 * hid + n];
    float go = bih[3 * hid + n] + bhh[3 * hid + n];
    const float* wi = Wih + n * in_dim;
    const float* wg = Wih + (2 * hid + n) * in_dim;
    const float* wo = Wih + (3 * hid + n) * in_dim;
    for (int k = 0; k < in_dim; ++k) {
      float xv = x[k];
      gi += xv * wi[k]; gg += xv * wg[k]; go += xv * wo[k];
    }
    float c = sigmoidf_(gi) * tanhf(gg);
    out[n] = sigmoidf_(go) * tanhf(c);
  }
}

__global__ void k_disc(const float* imputed,
                       const float* W0, const float* bi0, const float* bh0,
                       const float* W1, const float* bi1, const float* bh1,
                       const float* W2, const float* bi2, const float* bh2,
                       const float* W3, const float* bi3, const float* bh3,
                       const float* W4, const float* bi4, const float* bh4,
                       const float* doW, const float* dob, float* out) {
  int r = blockIdx.x * blockDim.x + threadIdx.x;
  if (r >= B_ * S_) return;
  float x0[8], buf1[32], buf2[32];
  #pragma unroll
  for (int d = 0; d < 8; ++d) x0[d] = imputed[r * 8 + d];
  disc_cell(W0, bi0, bh0, x0, 8, 32, buf1);
  disc_cell(W1, bi1, bh1, buf1, 32, 16, buf2);
  disc_cell(W2, bi2, bh2, buf2, 16, 8, buf1);
  disc_cell(W3, bi3, bh3, buf1, 8, 16, buf2);
  disc_cell(W4, bi4, bh4, buf2, 16, 32, buf1);
  for (int d = 0; d < 8; ++d) {
    float acc = dob[d];
    const float* w = doW + d * 32;
    for (int k = 0; k < 32; ++k) acc += buf1[k] * w[k];
    out[r * 8 + d] = acc;
  }
}

// decoder: batch-invariant -> one workgroup, 65 LSTM steps, outputs one [S,D] row
__global__ void __launch_bounds__(1024)
k_dec(const float* Wih, const float* Whh, const float* bih, const float* bhh,
      const float* Wdo, const float* bdo, float* recrow) {
  __shared__ float h[H_], c[H_], g[G4];
  int tid = threadIdx.x;
  if (tid < H_) { h[tid] = 0.0f; c[tid] = 0.0f; }
  __syncthreads();
  for (int step = 0; step <= S_; ++step) {
    #pragma unroll
    for (int jj = 0; jj < 2; ++jj) {
      int j = tid * 2 + jj;
      float acc = bih[j] + bhh[j];
      const float* wi = Wih + j * H_;
      const float* wh = Whh + j * H_;
      if (step == 0) {
        for (int k = 0; k < H_; ++k) acc += 128.0f * wi[k];   // h == 0
      } else {
        for (int k = 0; k < H_; ++k) acc += c[k] * wi[k] + h[k] * wh[k];
      }
      g[j] = acc;
    }
    __syncthreads();
    if (tid < H_) {
      float cn = sigmoidf_(g[H_ + tid]) * c[tid] + sigmoidf_(g[tid]) * tanhf(g[2 * H_ + tid]);
      c[tid] = cn;
      h[tid] = sigmoidf_(g[3 * H_ + tid]) * tanhf(cn);
    }
    __syncthreads();
    if (step >= 1 && tid < D_) {
      float acc = bdo[tid];
      const float* w = Wdo + tid * H_;
      for (int k = 0; k < H_; ++k) acc += h[k] * w[k];
      recrow[(step - 1) * D_ + tid] = acc;
    }
  }
}

__global__ void k_bcast(const float* recrow, float* out) {
  int i = blockIdx.x * blockDim.x + threadIdx.x;
  if (i < B_ * S_ * D_) out[i] = recrow[i & (S_ * D_ - 1)];
}

// ---------------- host launch ----------------

#define NB(n) (((n) + 255) / 256)

extern "C" void kernel_launch(void* const* d_in, const int* in_sizes, int n_in,
                              void* d_out, int out_size, void* d_ws, size_t ws_size,
                              hipStream_t stream) {
  (void)in_sizes; (void)n_in; (void)out_size; (void)ws_size;
  const float* values = (const float*)d_in[0];
  const int*   masks  = (const int*)d_in[1];
  // gen_fwd[0]=2..5, gen_fwd[1]=6..9, gen_bwd[0]=10..13, gen_bwd[1]=14..17
  const float* gf0_Wih = (const float*)d_in[2];
  const float* gf0_Whh = (const float*)d_in[3];
  const float* gf0_bih = (const float*)d_in[4];
  const float* gf0_bhh = (const float*)d_in[5];
  const float* gf1_Wih = (const float*)d_in[6];
  const float* gf1_Whh = (const float*)d_in[7];
  const float* gf1_bih = (const float*)d_in[8];
  const float* gf1_bhh = (const float*)d_in[9];
  const float* gb0_Wih = (const float*)d_in[10];
  const float* gb0_Whh = (const float*)d_in[11];
  const float* gb0_bih = (const float*)d_in[12];
  const float* gb0_bhh = (const float*)d_in[13];
  const float* gb1_Wih = (const float*)d_in[14];
  const float* gb1_Whh = (const float*)d_in[15];
  const float* gb1_bih = (const float*)d_in[16];
  const float* gb1_bhh = (const float*)d_in[17];
  const float* genout_W = (const float*)d_in[22];
  const float* genout_b = (const float*)d_in[23];
  const float* fc_W = (const float*)d_in[46];
  const float* fc_b = (const float*)d_in[47];
  const float* dec_Wih = (const float*)d_in[48];
  const float* dec_Whh = (const float*)d_in[49];
  const float* dec_bih = (const float*)d_in[50];
  const float* dec_bhh = (const float*)d_in[51];
  const float* deco_W = (const float*)d_in[52];
  const float* deco_b = (const float*)d_in[53];

  float* out = (float*)d_out;
  float* out_disc   = out;
  float* out_impute = out + 262144;
  float* out_lat    = out + 524288;
  float* out_rec    = out + 786432;

  char* wsb = (char*)d_ws;
  size_t off = 0;
  auto alloc = [&](size_t bytes) -> void* {
    off = (off + 255) & ~(size_t)255;
    void* p = wsb + off;
    off += bytes;
    return p;
  };
  u16* WCAT_F = (u16*)alloc((size_t)G4 * 1024 * 2);
  u16* WCAT_B = (u16*)alloc((size_t)G4 * 1024 * 2);
  u16* WHH0_F = (u16*)alloc((size_t)G4 * H_ * 2);
  u16* WHH0_B = (u16*)alloc((size_t)G4 * H_ * 2);
  u16* WIH0_F = (u16*)alloc((size_t)G4 * 32 * 2);
  u16* WIH0_B = (u16*)alloc((size_t)G4 * 32 * 2);
  u16* WFC    = (u16*)alloc((size_t)H_ * H_ * 2);
  float* B0F = (float*)alloc(G4 * 4);
  float* B0B = (float*)alloc(G4 * 4);
  float* B1F = (float*)alloc(G4 * 4);
  float* B1B = (float*)alloc(G4 * 4);
  u16* X1F = (u16*)alloc((size_t)B_ * 1024 * 2);
  u16* X1B = (u16*)alloc((size_t)B_ * 1024 * 2);
  float* C0F = (float*)alloc((size_t)B_ * H_ * 4);
  float* C0B = (float*)alloc((size_t)B_ * H_ * 4);
  float* C1F = (float*)alloc((size_t)B_ * H_ * 4);
  float* C1B = (float*)alloc((size_t)B_ * H_ * 4);
  float* HF  = (float*)alloc((size_t)B_ * H_ * 4);
  float* HB  = (float*)alloc((size_t)B_ * H_ * 4);
  float* GF  = (float*)alloc((size_t)B_ * G4 * 4);
  float* GB  = (float*)alloc((size_t)B_ * G4 * 4);
  u16* INF = (u16*)alloc((size_t)B_ * 32 * 2);
  u16* INB = (u16*)alloc((size_t)B_ * 32 * 2);
  float* WARMH = (float*)alloc(2 * H_ * 4);
  float* WARMC = (float*)alloc(2 * H_ * 4);
  u16* HBF = (u16*)alloc((size_t)B_ * H_ * 2);
  float* IMPUTED = (float*)alloc((size_t)B_ * H_ * 4);
  float* RECROW = (float*)alloc(S_ * D_ * 4);

  // ---- prep ----
  k_cat1<<<NB(G4 * 1024), 256, 0, stream>>>(WCAT_F, gf1_Wih, gf1_Whh);
  k_cat1<<<NB(G4 * 1024), 256, 0, stream>>>(WCAT_B, gb1_Wih, gb1_Whh);
  k_conv_bf<<<NB(G4 * H_), 256, 0, stream>>>(WHH0_F, gf0_Whh, G4 * H_);
  k_conv_bf<<<NB(G4 * H_), 256, 0, stream>>>(WHH0_B, gb0_Whh, G4 * H_);
  k_pad0<<<NB(G4 * 32), 256, 0, stream>>>(WIH0_F, gf0_Wih);
  k_pad0<<<NB(G4 * 32), 256, 0, stream>>>(WIH0_B, gb0_Wih);
  k_conv_bf<<<NB(H_ * H_), 256, 0, stream>>>(WFC, fc_W, H_ * H_);
  k_addbias<<<NB(G4), 256, 0, stream>>>(B0F, gf0_bih, gf0_bhh, G4);
  k_addbias<<<NB(G4), 256, 0, stream>>>(B0B, gb0_bih, gb0_bhh, G4);
  k_addbias<<<NB(G4), 256, 0, stream>>>(B1F, gf1_bih, gf1_bhh, G4);
  k_addbias<<<NB(G4), 256, 0, stream>>>(B1B, gb1_bih, gb1_bhh, G4);
  k_zero16<<<NB(B_ * 32), 256, 0, stream>>>(INF, B_ * 32);
  k_zero16<<<NB(B_ * 32), 256, 0, stream>>>(INB, B_ * 32);
  k_warm<<<NB(2 * H_), 256, 0, stream>>>(B1F, B1B, WARMH, WARMC);
  k_init<<<NB(2 * B_ * H_), 256, 0, stream>>>(WARMH, WARMC, X1F, X1B, C0F, C0B, C1F, C1B, HF, HB);

  // ---- recurrent scan: 64 timesteps, both directions per launch ----
  dim3 ggrid(16, 32, 2);   // M tiles 32 -> /2 ; N tiles 128 -> /4 ; 2 directions
  for (int t = 0; t < S_; ++t) {
    k_ximp<<<NB(2 * B_ * D_), 256, 0, stream>>>(HF, HB, genout_W, genout_b,
                                                values, masks, INF, INB, t);
    k_gemm_l0<<<ggrid, 256, 0, stream>>>(INF, INB, WIH0_F, WIH0_B,
                                         X1F, X1B, WHH0_F, WHH0_B, GF, GB);
    k_point<<<NB(2 * B_ * H_), 256, 0, stream>>>(GF, GB, B0F, B0B, C0F, C0B,
                                                 X1F, X1B, HF, HB, 0, 0);
    k_gemm_l1<<<ggrid, 256, 0, stream>>>(X1F, X1B, WCAT_F, WCAT_B, GF, GB);
    k_point<<<NB(2 * B_ * H_), 256, 0, stream>>>(GF, GB, B1F, B1B, C1F, C1B,
                                                 X1F, X1B, HF, HB, H_, 1);
  }

  // ---- merge, impute, latent, discriminator, decoder ----
  k_hmean<<<NB(B_ * H_), 256, 0, stream>>>(HF, HB, values, masks, out_impute, IMPUTED, HBF);
  k_gemm_fc<<<dim3(16, 8, 1), 256, 0, stream>>>(HBF, WFC, fc_b, out_lat);
  k_disc<<<NB(B_ * S_), 256, 0, stream>>>(IMPUTED,
      (const float*)d_in[24], (const float*)d_in[26], (const float*)d_in[27],
      (const float*)d_in[28], (const float*)d_in[30], (const float*)d_in[31],
      (const float*)d_in[32], (const float*)d_in[34], (const float*)d_in[35],
      (const float*)d_in[36], (const float*)d_in[38], (const float*)d_in[39],
      (const float*)d_in[40], (const float*)d_in[42], (const float*)d_in[43],
      (const float*)d_in[44], (const float*)d_in[45], out_disc);
  k_dec<<<1, 1024, 0, stream>>>(dec_Wih, dec_Whh, dec_bih, dec_bhh, deco_W, deco_b, RECROW);
  k_bcast<<<NB(B_ * S_ * D_), 256, 0, stream>>>(RECROW, out_rec);
}
